// FJSPNet_10445360464099
// MI455X (gfx1250) — matrix operand order repaired and checked by reading, not compiled, not run on hardware
//
#include <hip/hip_runtime.h>
#include <hip/hip_bf16.h>

typedef __bf16 bf16_t;
typedef __attribute__((ext_vector_type(16))) __bf16 v16bf;
typedef __attribute__((ext_vector_type(8)))  float  v8f;

#define NOPN 100000
#define NMN  1000
#define EEL  2000000
#define EPR  100000
#define NA   1000000
#define NLOG (NA + 1)
#define NRED 512

// ---- workspace layout (float offsets) ----
#define OFF_XOP   0LL
#define OFF_XM    6400000LL
#define OFF_AGGME 6464000LL
#define OFF_AGGPP 12864000LL
#define OFF_AGGM  19264000LL
#define OFF_CNTME 19328000LL
#define OFF_CNTPP 19428000LL
#define OFF_CNTM  19528000LL
#define ZERO_SPAN 13065000LL   // AGGME..CNTM contiguous
#define OFF_SUMOP 19529000LL
#define OFF_SUMM  19529064LL
#define OFF_PART  19529128LL
#define OFF_GMAX  19529640LL
#define OFF_GSUM  19529641LL
#define OFF_PACK  19529648LL   // bf16 region, 32B aligned
#define PKMAT(l, mi) (((l) * 5 + (mi)) * 4096)
#define PK_WP1 61440

// ---------------- WMMA helpers ----------------
__device__ __forceinline__ v8f wmma_bf16(v16bf a, v16bf b, v8f c) {
  return __builtin_amdgcn_wmma_f32_16x16x32_bf16(false, a, false, b, (short)0, c,
                                                 false, false);
}

// A fragment (16x32 bf16) from f32 row-major src (ld=64), per-row scale.
// lane: M = lane&15, K-offset 8 for hi half; element e -> K = e + (e&8).
__device__ __forceinline__ v16bf load_a(const float* __restrict__ src, int row,
                                        int kt, int hi, float scale) {
  const float* p = src + row * 64 + kt * 32 + hi * 8;
  v16bf a;
#pragma unroll
  for (int e = 0; e < 16; ++e) a[e] = (bf16_t)(p[e + (e & 8)] * scale);
  return a;
}

// B fragment (32x16 bf16) from pre-packed buffer: lane-contiguous 16 bf16.
__device__ __forceinline__ v16bf load_b(const bf16_t* __restrict__ p, int kt,
                                        int nt, int lane) {
  return *(const v16bf*)(p + ((kt * 4 + nt) * 32 + lane) * 16);
}

// ---------------- utility kernels ----------------
__global__ void k_zero(float* __restrict__ p, long long n) {
  long long i = (long long)blockIdx.x * blockDim.x + threadIdx.x;
  long long st = (long long)gridDim.x * blockDim.x;
  for (; i < n; i += st) p[i] = 0.f;
}

// pack 64x64 (and Wp1 128x64) f32 weights into bf16 WMMA B-fragment layout
__global__ void k_pack(const float* __restrict__ Wl_me, const float* __restrict__ Wl_pp,
                       const float* __restrict__ Wr_me, const float* __restrict__ Wr_pp,
                       const float* __restrict__ Wl_em, const float* __restrict__ Wr_em,
                       const float* __restrict__ Wp1, bf16_t* __restrict__ pack) {
  int tid = blockIdx.x * blockDim.x + threadIdx.x;
  const int total = 3 * 5 * 4096 + 8192;
  if (tid >= total) return;
  if (tid < PK_WP1) {
    int l = tid / (5 * 4096);
    int rem = tid - l * 5 * 4096;
    int mi = rem / 4096;
    int j = rem & 4095;
    int e = j & 15, lane = (j >> 4) & 31, nt = (j >> 9) & 3, kt = j >> 11;
    int k = kt * 32 + e + (e & 8) + ((lane >> 4) << 3);
    int n = nt * 16 + (lane & 15);
    int idx = l * 4096 + k * 64 + n;
    float v;
    if (mi == 0)      v = Wl_me[idx];
    else if (mi == 1) v = Wl_pp[idx];
    else if (mi == 2) v = Wr_me[idx] + Wr_pp[idx];
    else if (mi == 3) v = Wl_em[idx];
    else              v = Wr_em[idx];
    pack[tid] = (bf16_t)v;
  } else {
    int j = tid - PK_WP1;
    int e = j & 15, lane = (j >> 4) & 31, nt = (j >> 9) & 3, kt = j >> 11;
    int k = kt * 32 + e + (e & 8) + ((lane >> 4) << 3);
    int n = nt * 16 + (lane & 15);
    pack[tid] = (bf16_t)Wp1[k * 64 + n];
  }
}

// input projection: 64 threads per row, relu(x @ W + b)
__global__ void k_proj(const float* __restrict__ xin, const float* __restrict__ W,
                       const float* __restrict__ b, float* __restrict__ xout,
                       int nrows, int din) {
  int t = blockIdx.x * blockDim.x + threadIdx.x;
  int row = t >> 6, c = t & 63;
  if (row >= nrows) return;
  float acc = b[c];
  for (int d = 0; d < din; ++d) acc += xin[row * din + d] * W[d * 64 + c];
  xout[row * 64 + c] = fmaxf(acc, 0.f);
}

// eligible edges: both directions in one pass; 64 lanes per edge (coalesced)
__global__ void k_scatter_elig(const int* __restrict__ src, const int* __restrict__ dst,
                               const float* __restrict__ x_op, const float* __restrict__ x_m,
                               float* __restrict__ agg_m, float* __restrict__ cnt_m,
                               float* __restrict__ agg_op, float* __restrict__ cnt_op,
                               int ne) {
  int tid = blockIdx.x * blockDim.x + threadIdx.x;
  int e = tid >> 6, c = tid & 63;
  if (e >= ne) return;
  int s = src[e], d = dst[e];
  atomicAdd(&agg_m[d * 64 + c], x_op[s * 64 + c]);
  atomicAdd(&agg_op[s * 64 + c], x_m[d * 64 + c]);
  if (c == 0) { atomicAdd(&cnt_m[d], 1.f); atomicAdd(&cnt_op[s], 1.f); }
}

__global__ void k_scatter_pp(const int* __restrict__ src, const int* __restrict__ dst,
                             const float* __restrict__ x_op,
                             float* __restrict__ agg, float* __restrict__ cnt, int ne) {
  int tid = blockIdx.x * blockDim.x + threadIdx.x;
  int e = tid >> 6, c = tid & 63;
  if (e >= ne) return;
  int s = src[e], d = dst[e];
  atomicAdd(&agg[d * 64 + c], x_op[s * 64 + c]);
  if (c == 0) atomicAdd(&cnt[d], 1.f);
}

// Node update: x = LN(x + mean1@B1 [+ mean2@B2] + x@Bself + biases), in place.
// One wave per 16-row tile, WMMA bf16 16x16x32, f32 accum, half-wave LN reduce.
__global__ void k_node_update(float* __restrict__ x, int nrows,
                              const float* __restrict__ agg1, const float* __restrict__ cnt1,
                              const bf16_t* __restrict__ B1,
                              const float* __restrict__ agg2, const float* __restrict__ cnt2,
                              const bf16_t* __restrict__ B2,
                              const bf16_t* __restrict__ Bself,
                              const float* __restrict__ bias1, const float* __restrict__ bias2,
                              const float* __restrict__ lng, const float* __restrict__ lnb,
                              int ntiles) {
  int wid = (blockIdx.x * blockDim.x + threadIdx.x) >> 5;
  if (wid >= ntiles) return;
  int lane = threadIdx.x & 31;
  int m = lane & 15, hi = lane >> 4, n0 = lane & 15;
  int base = wid * 16;
  int arow = base + m; if (arow > nrows - 1) arow = nrows - 1;

  float inv1 = 1.f / fmaxf(cnt1[arow], 1.f);
  v16bf a1[2], as[2], a2[2];
  a1[0] = load_a(agg1, arow, 0, hi, inv1);
  a1[1] = load_a(agg1, arow, 1, hi, inv1);
  as[0] = load_a(x, arow, 0, hi, 1.f);
  as[1] = load_a(x, arow, 1, hi, 1.f);
  bool has2 = (agg2 != nullptr);
  if (has2) {
    float inv2 = 1.f / fmaxf(cnt2[arow], 1.f);
    a2[0] = load_a(agg2, arow, 0, hi, inv2);
    a2[1] = load_a(agg2, arow, 1, hi, inv2);
  }

  float v[4][8];
#pragma unroll
  for (int t = 0; t < 4; ++t) {
    int col = t * 16 + n0;
    float bias = bias1[col] + (bias2 ? bias2[col] : 0.f);
    v8f c;
#pragma unroll
    for (int r = 0; r < 8; ++r) c[r] = bias;
#pragma unroll
    for (int kt = 0; kt < 2; ++kt) c = wmma_bf16(a1[kt], load_b(B1, kt, t, lane), c);
    if (has2) {
#pragma unroll
      for (int kt = 0; kt < 2; ++kt) c = wmma_bf16(a2[kt], load_b(B2, kt, t, lane), c);
    }
#pragma unroll
    for (int kt = 0; kt < 2; ++kt) c = wmma_bf16(as[kt], load_b(Bself, kt, t, lane), c);
#pragma unroll
    for (int r = 0; r < 8; ++r) {
      int row = base + r + 8 * hi;
      int rr = row > nrows - 1 ? nrows - 1 : row;
      v[t][r] = c[r] + x[rr * 64 + col];   // residual
    }
  }

  // per-row LayerNorm over 64 cols distributed across 16 lanes x 4 tiles
  float mu[8], rs[8];
#pragma unroll
  for (int r = 0; r < 8; ++r) {
    float s = v[0][r] + v[1][r] + v[2][r] + v[3][r];
    float q = v[0][r] * v[0][r] + v[1][r] * v[1][r] + v[2][r] * v[2][r] + v[3][r] * v[3][r];
#pragma unroll
    for (int msk = 1; msk < 16; msk <<= 1) {
      s += __shfl_xor(s, msk, 16);
      q += __shfl_xor(q, msk, 16);
    }
    mu[r] = s * (1.f / 64.f);
    float var = q * (1.f / 64.f) - mu[r] * mu[r];
    rs[r] = rsqrtf(var + 1e-5f);
  }
#pragma unroll
  for (int t = 0; t < 4; ++t) {
    int col = t * 16 + n0;
    float g = lng[col], bb = lnb[col];
#pragma unroll
    for (int r = 0; r < 8; ++r) {
      int row = base + r + 8 * hi;
      if (row < nrows) x[row * 64 + col] = (v[t][r] - mu[r]) * rs[r] * g + bb;
    }
  }
}

// Policy head: per-wave 16 actions; gather [x_op|x_m] 128 dims -> 4 WMMA K-tiles;
// tm (2 dims) + bias folded into C init; relu; dot Wp2 via half-wave reduce.
__global__ void k_policy(const float* __restrict__ x_op, const float* __restrict__ x_m,
                         const float* __restrict__ tmf,
                         const int* __restrict__ aop, const int* __restrict__ am,
                         const int* __restrict__ atm, const unsigned char* __restrict__ mask,
                         const bf16_t* __restrict__ Wp1p, const float* __restrict__ bp1,
                         const float* __restrict__ Wp1, const float* __restrict__ wp2,
                         const float* __restrict__ bp2, float* __restrict__ logits,
                         int ntiles) {
  int wid = (blockIdx.x * blockDim.x + threadIdx.x) >> 5;
  if (wid >= ntiles) return;
  int lane = threadIdx.x & 31;
  int m = lane & 15, hi = lane >> 4, n0 = lane & 15;
  int base = wid * 16;
  int arow = base + m;
  int iop = aop[arow], im = am[arow], itm = atm[arow];
  float tm0 = tmf[itm * 2], tm1 = tmf[itm * 2 + 1];
  const float* po = x_op + (long long)iop * 64;
  const float* pm = x_m + (long long)im * 64;

  v16bf a[4];
#pragma unroll
  for (int kt = 0; kt < 4; ++kt) {
#pragma unroll
    for (int e = 0; e < 16; ++e) {
      int k = kt * 32 + e + (e & 8) + hi * 8;
      a[kt][e] = (bf16_t)(k < 64 ? po[k] : pm[k - 64]);
    }
  }
  float t0s[8], t1s[8];
#pragma unroll
  for (int r = 0; r < 8; ++r) {
    int rr = r + 8 * hi;
    t0s[r] = __shfl(tm0, rr, 32);
    t1s[r] = __shfl(tm1, rr, 32);
  }
  float part[8];
#pragma unroll
  for (int r = 0; r < 8; ++r) part[r] = 0.f;
#pragma unroll
  for (int t = 0; t < 4; ++t) {
    int col = t * 16 + n0;
    float w128 = Wp1[128 * 64 + col], w129 = Wp1[129 * 64 + col], bb = bp1[col];
    v8f c;
#pragma unroll
    for (int r = 0; r < 8; ++r) c[r] = bb + t0s[r] * w128 + t1s[r] * w129;
#pragma unroll
    for (int kt = 0; kt < 4; ++kt) c = wmma_bf16(a[kt], load_b(Wp1p, kt, t, lane), c);
    float w2 = wp2[col];
#pragma unroll
    for (int r = 0; r < 8; ++r) part[r] += fmaxf(c[r], 0.f) * w2;
  }
#pragma unroll
  for (int r = 0; r < 8; ++r) {
#pragma unroll
    for (int msk = 1; msk < 16; msk <<= 1) part[r] += __shfl_xor(part[r], msk, 16);
  }
  if (n0 == 0) {
    float b2 = bp2[0];
#pragma unroll
    for (int r = 0; r < 8; ++r) {
      int g = base + r + 8 * hi;
      float lg = part[r] + b2;
      logits[g] = mask[g] ? lg : -__builtin_inff();
    }
  }
}

__global__ void k_colsum(const float* __restrict__ x, int nrows, float* __restrict__ out) {
  int tid = blockIdx.x * blockDim.x + threadIdx.x;
  int c = tid & 63, g = tid >> 6;
  int stride = (gridDim.x * blockDim.x) >> 6;
  float s = 0.f;
  for (int r = g; r < nrows; r += stride) s += x[r * 64 + c];
  atomicAdd(&out[c], s);
}

__global__ void k_value_idle(const float* __restrict__ sum_op, const float* __restrict__ sum_m,
                             const float* __restrict__ gf,
                             const float* __restrict__ Wv1, const float* __restrict__ bv1,
                             const float* __restrict__ Wv2, const float* __restrict__ bv2,
                             const float* __restrict__ idle_emb, const float* __restrict__ Wp1,
                             const float* __restrict__ bp1, const float* __restrict__ Wp2,
                             const float* __restrict__ bp2, const unsigned char* __restrict__ mask,
                             float* __restrict__ out) {
  __shared__ float vin[133];
  __shared__ float hv[64], hidle[64];
  int t = threadIdx.x;
  if (t < 64) vin[t] = sum_op[t] * (1.f / (float)NOPN);
  else if (t < 128) vin[t] = sum_m[t - 64] * (1.f / (float)NMN);
  else if (t < 133) vin[t] = gf[t - 128];
  __syncthreads();
  if (t < 64) {
    float a = bv1[t];
    for (int k = 0; k < 133; ++k) a += vin[k] * Wv1[k * 64 + t];
    hv[t] = fmaxf(a, 0.f);
    float b = bp1[t];
    for (int k = 0; k < 130; ++k) b += idle_emb[k] * Wp1[k * 64 + t];
    hidle[t] = fmaxf(b, 0.f);
  }
  __syncthreads();
  if (t == 0) {
    float a = bv2[0], b = bp2[0];
    for (int k = 0; k < 64; ++k) { a += hv[k] * Wv2[k]; b += hidle[k] * Wp2[k]; }
    out[NA + 1] = tanhf(a);                                   // value
    out[NA] = mask[NA] ? b : -__builtin_inff();               // idle logit
  }
}

// ---- softmax (3 passes, two-stage reductions) ----
__global__ void k_max1(const float* __restrict__ x, int n, float* __restrict__ partial) {
  __shared__ float s[256];
  float mv = -__builtin_inff();
  for (int i = blockIdx.x * blockDim.x + threadIdx.x; i < n; i += gridDim.x * blockDim.x)
    mv = fmaxf(mv, x[i]);
  s[threadIdx.x] = mv; __syncthreads();
  for (int off = 128; off > 0; off >>= 1) {
    if (threadIdx.x < off) s[threadIdx.x] = fmaxf(s[threadIdx.x], s[threadIdx.x + off]);
    __syncthreads();
  }
  if (threadIdx.x == 0) partial[blockIdx.x] = s[0];
}
__global__ void k_max2(const float* __restrict__ partial, int np, float* __restrict__ gmax) {
  __shared__ float s[256];
  float mv = -__builtin_inff();
  for (int i = threadIdx.x; i < np; i += 256) mv = fmaxf(mv, partial[i]);
  s[threadIdx.x] = mv; __syncthreads();
  for (int off = 128; off > 0; off >>= 1) {
    if (threadIdx.x < off) s[threadIdx.x] = fmaxf(s[threadIdx.x], s[threadIdx.x + off]);
    __syncthreads();
  }
  if (threadIdx.x == 0) gmax[0] = s[0];
}
__global__ void k_sum1(const float* __restrict__ x, int n, const float* __restrict__ gmax,
                       float* __restrict__ partial) {
  __shared__ float s[256];
  float gm = gmax[0], acc = 0.f;
  for (int i = blockIdx.x * blockDim.x + threadIdx.x; i < n; i += gridDim.x * blockDim.x)
    acc += __expf(x[i] - gm);
  s[threadIdx.x] = acc; __syncthreads();
  for (int off = 128; off > 0; off >>= 1) {
    if (threadIdx.x < off) s[threadIdx.x] += s[threadIdx.x + off];
    __syncthreads();
  }
  if (threadIdx.x == 0) partial[blockIdx.x] = s[0];
}
__global__ void k_sum2(const float* __restrict__ partial, int np, float* __restrict__ gsum) {
  __shared__ float s[256];
  float acc = 0.f;
  for (int i = threadIdx.x; i < np; i += 256) acc += partial[i];
  s[threadIdx.x] = acc; __syncthreads();
  for (int off = 128; off > 0; off >>= 1) {
    if (threadIdx.x < off) s[threadIdx.x] += s[threadIdx.x + off];
    __syncthreads();
  }
  if (threadIdx.x == 0) gsum[0] = s[0];
}
__global__ void k_norm(float* __restrict__ x, int n, const float* __restrict__ gmax,
                       const float* __restrict__ gsum) {
  float gm = gmax[0], inv = 1.f / gsum[0];
  for (int i = blockIdx.x * blockDim.x + threadIdx.x; i < n; i += gridDim.x * blockDim.x)
    x[i] = __expf(x[i] - gm) * inv;
}

extern "C" void kernel_launch(void* const* d_in, const int* in_sizes, int n_in,
                              void* d_out, int out_size, void* d_ws, size_t ws_size,
                              hipStream_t stream) {
  const float* op_x      = (const float*)d_in[0];
  const float* machine_x = (const float*)d_in[1];
  const float* gfeat     = (const float*)d_in[2];
  const int*   elig_src  = (const int*)d_in[3];
  const int*   elig_dst  = (const int*)d_in[4];
  const int*   prec_src  = (const int*)d_in[5];
  const int*   prec_dst  = (const int*)d_in[6];
  const int*   aop       = (const int*)d_in[7];
  const int*   am        = (const int*)d_in[8];
  const int*   atm       = (const int*)d_in[9];
  const unsigned char* mask = (const unsigned char*)d_in[10];
  const float* tmf  = (const float*)d_in[11];
  const float* Wop  = (const float*)d_in[12];
  const float* bop  = (const float*)d_in[13];
  const float* Wm   = (const float*)d_in[14];
  const float* bm   = (const float*)d_in[15];
  const float* Wl_em = (const float*)d_in[16];
  const float* bl_em = (const float*)d_in[17];
  const float* Wr_em = (const float*)d_in[18];
  const float* Wl_me = (const float*)d_in[19];
  const float* bl_me = (const float*)d_in[20];
  const float* Wr_me = (const float*)d_in[21];
  const float* Wl_pp = (const float*)d_in[22];
  const float* bl_pp = (const float*)d_in[23];
  const float* Wr_pp = (const float*)d_in[24];
  const float* ln_op_g = (const float*)d_in[25];
  const float* ln_op_b = (const float*)d_in[26];
  const float* ln_m_g  = (const float*)d_in[27];
  const float* ln_m_b  = (const float*)d_in[28];
  const float* Wp1 = (const float*)d_in[29];
  const float* bp1 = (const float*)d_in[30];
  const float* Wp2 = (const float*)d_in[31];
  const float* bp2 = (const float*)d_in[32];
  const float* idle_emb = (const float*)d_in[33];
  const float* Wv1 = (const float*)d_in[34];
  const float* bv1 = (const float*)d_in[35];
  const float* Wv2 = (const float*)d_in[36];
  const float* bv2 = (const float*)d_in[37];

  float* W = (float*)d_ws;
  float* x_op   = W + OFF_XOP;
  float* x_m    = W + OFF_XM;
  float* agg_me = W + OFF_AGGME;
  float* agg_pp = W + OFF_AGGPP;
  float* agg_m  = W + OFF_AGGM;
  float* cnt_me = W + OFF_CNTME;
  float* cnt_pp = W + OFF_CNTPP;
  float* cnt_m  = W + OFF_CNTM;
  float* sum_op = W + OFF_SUMOP;
  float* sum_m  = W + OFF_SUMM;
  float* part   = W + OFF_PART;
  float* gmax   = W + OFF_GMAX;
  float* gsum   = W + OFF_GSUM;
  bf16_t* pack  = (bf16_t*)(W + OFF_PACK);
  float* out    = (float*)d_out;

  // pack weights into WMMA B-fragment layout (bf16)
  k_pack<<<(3 * 5 * 4096 + 8192 + 255) / 256, 256, 0, stream>>>(
      Wl_me, Wl_pp, Wr_me, Wr_pp, Wl_em, Wr_em, Wp1, pack);

  // input projections
  k_proj<<<(NOPN * 64) / 256, 256, 0, stream>>>(op_x, Wop, bop, x_op, NOPN, 10);
  k_proj<<<(NMN * 64) / 256, 256, 0, stream>>>(machine_x, Wm, bm, x_m, NMN, 6);

  for (int l = 0; l < 3; ++l) {
    k_zero<<<2048, 256, 0, stream>>>(W + OFF_AGGME, ZERO_SPAN);
    k_scatter_elig<<<(EEL * 64) / 256, 256, 0, stream>>>(
        elig_src, elig_dst, x_op, x_m, agg_m, cnt_m, agg_me, cnt_me, EEL);
    k_scatter_pp<<<(EPR * 64) / 256, 256, 0, stream>>>(
        prec_src, prec_dst, x_op, agg_pp, cnt_pp, EPR);
    // op update: mean_me@Wl_me + mean_pp@Wl_pp + x_op@(Wr_me+Wr_pp) + biases, +res, LN
    int op_tiles = NOPN / 16;
    k_node_update<<<(op_tiles * 32 + 255) / 256, 256, 0, stream>>>(
        x_op, NOPN, agg_me, cnt_me, pack + PKMAT(l, 0), agg_pp, cnt_pp,
        pack + PKMAT(l, 1), pack + PKMAT(l, 2), bl_me + l * 64, bl_pp + l * 64,
        ln_op_g + l * 64, ln_op_b + l * 64, op_tiles);
    // machine update: mean_m@Wl_em + x_m@Wr_em + bl_em, +res, LN
    int m_tiles = (NMN + 15) / 16;
    k_node_update<<<(m_tiles * 32 + 255) / 256, 256, 0, stream>>>(
        x_m, NMN, agg_m, cnt_m, pack + PKMAT(l, 3), nullptr, nullptr, nullptr,
        pack + PKMAT(l, 4), bl_em + l * 64, nullptr, ln_m_g + l * 64,
        ln_m_b + l * 64, m_tiles);
  }

  // value head inputs
  k_zero<<<1, 128, 0, stream>>>(sum_op, 128);
  k_colsum<<<256, 256, 0, stream>>>(x_op, NOPN, sum_op);
  k_colsum<<<256, 256, 0, stream>>>(x_m, NMN, sum_m);
  k_value_idle<<<1, 256, 0, stream>>>(sum_op, sum_m, gfeat, Wv1, bv1, Wv2, bv2,
                                      idle_emb, Wp1, bp1, Wp2, bp2, mask, out);

  // policy logits
  int p_tiles = NA / 16;
  k_policy<<<(p_tiles * 32 + 255) / 256, 256, 0, stream>>>(
      x_op, x_m, tmf, aop, am, atm, mask, pack + PK_WP1, bp1, Wp1, Wp2, bp2,
      out, p_tiles);

  // softmax over NA+1 logits (value at NA+1 untouched)
  k_max1<<<NRED, 256, 0, stream>>>(out, NLOG, part);
  k_max2<<<1, 256, 0, stream>>>(part, NRED, gmax);
  k_sum1<<<NRED, 256, 0, stream>>>(out, NLOG, gmax, part);
  k_sum2<<<1, 256, 0, stream>>>(part, NRED, gsum);
  k_norm<<<2048, 256, 0, stream>>>(out, NLOG, gmax, gsum);
}
